// MoE_AllToAll_Layer_73735998538236
// MI455X (gfx1250) — matrix operand order repaired and checked by reading, not compiled, not run on hardware
//
#include <hip/hip_runtime.h>
#include <hip/hip_bf16.h>

#define N_TOKENS 4096
#define D_IN     1024
#define D_HID    4096
#define D_OUT    1024
#define N_EXP    8

#define BM 128
#define BN 128
#define BK 32
#define BKP 40              // padded K pitch (ushorts) for transposed B tile
#define MAX_TILES 40        // ceil(4096/128) + 8 partial tiles

typedef __attribute__((ext_vector_type(16))) __bf16 v16bf;
typedef __attribute__((ext_vector_type(8)))  float  v8f;
typedef __attribute__((ext_vector_type(4)))  unsigned int u32x4;

union FragCvt { u32x4 q[2]; v16bf v; };

// Pack two fp32 -> two bf16 (round-half-up) in one dword with a single v_perm_b32.
__device__ __forceinline__ unsigned int pack2bf(float a, float b) {
  const unsigned int ua = __float_as_uint(a) + 0x8000u;
  const unsigned int ub = __float_as_uint(b) + 0x8000u;
  // dst[15:0] = ua[31:16] (src1 bytes 2,3), dst[31:16] = ub[31:16] (src0 bytes 6,7)
  return __builtin_amdgcn_perm(ub, ua, 0x07060302u);
}

__device__ __forceinline__ unsigned short f2bf(float f) {
  unsigned int u = __float_as_uint(f);
  u += 0x7fffu + ((u >> 16) & 1u);
  return (unsigned short)(u >> 16);
}

// A tile in LDS: [BM][BK] bf16 row-major. A fragment k order used here (same as B):
// lane l -> row = l%16, K = (l/16)*16 .. +15 contiguous (matches B).
__device__ __forceinline__ v16bf load_a_frag(const unsigned short* A, int mi, int lane) {
  const int row = mi * 16 + (lane & 15);
  const int kl  = (lane >> 4) * 16;
  FragCvt f;
  f.q[0] = *(const u32x4*)(A + row * BK + kl);
  f.q[1] = *(const u32x4*)(A + row * BK + kl + 8);
  return f.v;
}

// B tile in LDS stored transposed: [BN][BKP] bf16. B fragment k order used here (same as A):
// lane l -> col n = l%16, K = (l/16)*16 .. +15 contiguous.
__device__ __forceinline__ v16bf load_b_frag(const unsigned short* B, int ni, int lane) {
  const int n  = ni * 16 + (lane & 15);
  const int kh = (lane >> 4) * 16;
  const u32x4* p = (const u32x4*)(B + n * BKP + kh);
  FragCvt f;
  f.q[0] = p[0];
  f.q[1] = p[1];
  return f.v;
}

// ---------------- Router: one wave per token ----------------
__global__ __launch_bounds__(256)
void moe_router(const float* __restrict__ x, const float* __restrict__ Wr,
                const float* __restrict__ br, int* __restrict__ topi,
                float* __restrict__ tops) {
  const int wave = threadIdx.x >> 5;
  const int lane = threadIdx.x & 31;
  const int t = blockIdx.x * 8 + wave;
  if (t >= N_TOKENS) return;

  float acc[N_EXP];
#pragma unroll
  for (int e = 0; e < N_EXP; ++e) acc[e] = 0.f;

  const float* xr = x + (size_t)t * D_IN;
  for (int k = lane; k < D_IN; k += 32) {
    const float xv = xr[k];
    const float* w = Wr + (size_t)k * N_EXP;
#pragma unroll
    for (int e = 0; e < N_EXP; ++e) acc[e] += xv * w[e];
  }
#pragma unroll
  for (int off = 16; off > 0; off >>= 1) {
#pragma unroll
    for (int e = 0; e < N_EXP; ++e) acc[e] += __shfl_xor(acc[e], off, 32);
  }
  if (lane == 0) {
    float m = -3.4e38f; int mi = 0;
#pragma unroll
    for (int e = 0; e < N_EXP; ++e) {
      const float l = acc[e] + br[e];
      acc[e] = l;
      if (l > m) { m = l; mi = e; }     // first index on tie (argmax semantics)
    }
    float s = 0.f;
#pragma unroll
    for (int e = 0; e < N_EXP; ++e) s += __expf(acc[e] - m);
    tops[t] = 1.0f / s;                  // max softmax score
    topi[t] = mi;
  }
}

// ------------- Stable counting sort + tile list (1 block) -------------
__global__ __launch_bounds__(256)
void moe_sort(const int* __restrict__ topi, const float* __restrict__ tops,
              int* __restrict__ perm, float* __restrict__ scal,
              int* __restrict__ te, int* __restrict__ tr, int* __restrict__ tn) {
  __shared__ int cnt[256][N_EXP];
  __shared__ int tot[N_EXP];
  __shared__ int base[N_EXP + 1];
  const int tid = threadIdx.x;
  const int CH = N_TOKENS / 256;   // 16 tokens per thread
  const int start = tid * CH;

  int myTop[CH];
  int lc[N_EXP];
#pragma unroll
  for (int e = 0; e < N_EXP; ++e) lc[e] = 0;
  for (int i = 0; i < CH; ++i) { myTop[i] = topi[start + i]; lc[myTop[i]]++; }
#pragma unroll
  for (int e = 0; e < N_EXP; ++e) cnt[tid][e] = lc[e];
  __syncthreads();

  if (tid < N_EXP) {                 // exclusive prefix over chunks, per expert
    int run = 0;
    for (int c = 0; c < 256; ++c) { const int v = cnt[c][tid]; cnt[c][tid] = run; run += v; }
    tot[tid] = run;
  }
  __syncthreads();

  if (tid == 0) {
    base[0] = 0;
    for (int e = 0; e < N_EXP; ++e) base[e + 1] = base[e] + tot[e];
    int nt = 0;                      // build padded-tile list per expert
    for (int e = 0; e < N_EXP; ++e) {
      for (int s = base[e]; s < base[e + 1]; s += BM) {
        const int rem = base[e + 1] - s;
        te[nt] = e; tr[nt] = s; tn[nt] = rem < BM ? rem : BM; ++nt;
      }
    }
    for (; nt < MAX_TILES; ++nt) { te[nt] = 0; tr[nt] = 0; tn[nt] = 0; }
  }
  __syncthreads();

  int r[N_EXP];
#pragma unroll
  for (int e = 0; e < N_EXP; ++e) r[e] = cnt[tid][e];
  for (int i = 0; i < CH; ++i) {     // stable scatter
    const int e = myTop[i];
    perm[base[e] + r[e]] = start + i;
    r[e]++;
  }
  __syncthreads();
  for (int i = 0; i < CH; ++i) {     // out row j scaled by score of j-th sorted token
    const int j = start + i;
    scal[j] = tops[perm[j]];
  }
}

// ------------- GEMM1: gather(x) @ W1[e] + b1 -> relu -> h (bf16) -------------
__global__ __launch_bounds__(256)
void moe_ffn1(const float* __restrict__ x, const float* __restrict__ W1,
              const float* __restrict__ b1, const int* __restrict__ perm,
              const int* __restrict__ te, const int* __restrict__ tr,
              const int* __restrict__ tn, unsigned short* __restrict__ h) {
  __shared__ __align__(16) unsigned short Als[2][BM * BK];
  __shared__ __align__(16) unsigned short Bls[2][BN * BKP];

  const int tile = blockIdx.x;
  const int rows = tn[tile];
  if (rows == 0) return;
  const int e    = te[tile];
  const int row0 = tr[tile];
  const int n0   = blockIdx.y * BN;
  const float* W = W1 + (size_t)e * D_IN * D_HID;

  const int tid  = threadIdx.x;
  const int lane = tid & 31;
  const int wave = tid >> 5;
  const int mi0  = (wave & 3) * 2;   // 2 M-tiles per wave
  const int ni0  = (wave >> 2) * 4;  // 4 N-tiles per wave

  v8f acc[2][4] = {};

  // ---- A loader: thread -> (row, 16-wide K segment) ----
  const int arow = tid >> 1;
  const int acol = (tid & 1) * 16;
  const float* asrc = (arow < rows) ? (x + (size_t)perm[row0 + arow] * D_IN + acol)
                                    : (const float*)0;
  const int aoff = arow * BK + acol;
  if (!asrc) {      // zero padded rows once (slots never rewritten below)
#pragma unroll
    for (int i = 0; i < 8; ++i) {
      ((unsigned int*)(Als[0] + aoff))[i] = 0u;
      ((unsigned int*)(Als[1] + aoff))[i] = 0u;
    }
  }

  // ---- B loader: thread -> (2 K-rows, 8 N) so (k,k+1) pairs pack to b32 ----
  const int kp = (tid >> 4) * 2;        // 0,2,..,30
  const int nsub = (tid & 15) * 8;      // 0..120
  const float* bsrc0 = W + (size_t)kp * D_HID + n0 + nsub;

  int cur = 0;
  // prologue: stage k0 = 0 into buffer 0
  if (asrc) {
#pragma unroll
    for (int i = 0; i < 8; ++i)
      ((unsigned int*)(Als[0] + aoff))[i] = pack2bf(asrc[2 * i], asrc[2 * i + 1]);
  }
  {
    const float* brow = bsrc0;
#pragma unroll
    for (int i = 0; i < 8; ++i)
      *(unsigned int*)(Bls[0] + (nsub + i) * BKP + kp) = pack2bf(brow[i], brow[i + D_HID]);
  }

  for (int k0 = 0; k0 < D_IN; k0 += BK) {
    __syncthreads();                      // buffer 'cur' is ready
    const int kn = k0 + BK;
    if (kn < D_IN) {                      // stage next tile into 'cur^1'
      if (asrc) {
#pragma unroll
        for (int i = 0; i < 8; ++i)
          ((unsigned int*)(Als[cur ^ 1] + aoff))[i] =
              pack2bf(asrc[kn + 2 * i], asrc[kn + 2 * i + 1]);
      }
      const float* brow = bsrc0 + (size_t)kn * D_HID;
      if (kn + BK < D_IN) __builtin_prefetch(brow + (size_t)BK * D_HID, 0, 0);
#pragma unroll
      for (int i = 0; i < 8; ++i)
        *(unsigned int*)(Bls[cur ^ 1] + (nsub + i) * BKP + kp) =
            pack2bf(brow[i], brow[i + D_HID]);
    }
#pragma unroll
    for (int m = 0; m < 2; ++m) {
      const v16bf af = load_a_frag(Als[cur], mi0 + m, lane);
#pragma unroll
      for (int n = 0; n < 4; ++n) {
        const v16bf bfr = load_b_frag(Bls[cur], ni0 + n, lane);
        acc[m][n] = __builtin_amdgcn_wmma_f32_16x16x32_bf16(
            false, af, false, bfr, (short)0, acc[m][n], false, false);
      }
    }
    cur ^= 1;
  }

  const float* bias = b1 + (size_t)e * D_HID + n0;
  const int lcol  = lane & 15;
  const int rhalf = (lane >> 4) * 8;   // C/D layout: M = v + 8*(lane/16)
  float bv[4];
#pragma unroll
  for (int n = 0; n < 4; ++n) bv[n] = bias[(ni0 + n) * 16 + lcol];
#pragma unroll
  for (int m = 0; m < 2; ++m) {
#pragma unroll
    for (int v = 0; v < 8; ++v) {
      const int rloc = (mi0 + m) * 16 + rhalf + v;
      if (rloc < rows) {
        unsigned short* hrow = h + (size_t)(row0 + rloc) * D_HID + n0;
#pragma unroll
        for (int n = 0; n < 4; ++n) {
          float o = acc[m][n][v] + bv[n];
          o = o > 0.f ? o : 0.f;
          hrow[(ni0 + n) * 16 + lcol] = f2bf(o);
        }
      }
    }
  }
}

// ------------- GEMM2: h @ W2[e] + b2, scale, scatter to token rows -------------
__global__ __launch_bounds__(256)
void moe_ffn2(const unsigned short* __restrict__ h, const float* __restrict__ W2,
              const float* __restrict__ b2, const int* __restrict__ perm,
              const float* __restrict__ scal,
              const int* __restrict__ te, const int* __restrict__ tr,
              const int* __restrict__ tn, float* __restrict__ out) {
  __shared__ __align__(16) unsigned short Als[2][BM * BK];
  __shared__ __align__(16) unsigned short Bls[2][BN * BKP];

  const int tile = blockIdx.x;
  const int rows = tn[tile];
  if (rows == 0) return;
  const int e    = te[tile];
  const int row0 = tr[tile];
  const int n0   = blockIdx.y * BN;
  const float* W = W2 + (size_t)e * D_HID * D_OUT;

  const int tid  = threadIdx.x;
  const int lane = tid & 31;
  const int wave = tid >> 5;
  const int mi0  = (wave & 3) * 2;
  const int ni0  = (wave >> 2) * 4;

  v8f acc[2][4] = {};

  const int arow = tid >> 1;
  const int acol = (tid & 1) * 16;
  const unsigned short* asrc =
      (arow < rows) ? (h + (size_t)(row0 + arow) * D_HID + acol)
                    : (const unsigned short*)0;
  const int aoff = arow * BK + acol;
  if (!asrc) {
    const u32x4 z = {};
    ((u32x4*)(Als[0] + aoff))[0] = z; ((u32x4*)(Als[0] + aoff))[1] = z;
    ((u32x4*)(Als[1] + aoff))[0] = z; ((u32x4*)(Als[1] + aoff))[1] = z;
  }

  const int kp = (tid >> 4) * 2;
  const int nsub = (tid & 15) * 8;
  const float* bsrc0 = W + (size_t)kp * D_OUT + n0 + nsub;

  int cur = 0;
  if (asrc) {
    const u32x4* s = (const u32x4*)asrc;
    ((u32x4*)(Als[0] + aoff))[0] = s[0];
    ((u32x4*)(Als[0] + aoff))[1] = s[1];
  }
  {
    const float* brow = bsrc0;
#pragma unroll
    for (int i = 0; i < 8; ++i)
      *(unsigned int*)(Bls[0] + (nsub + i) * BKP + kp) = pack2bf(brow[i], brow[i + D_OUT]);
  }

  for (int k0 = 0; k0 < D_HID; k0 += BK) {
    __syncthreads();
    const int kn = k0 + BK;
    if (kn < D_HID) {
      if (asrc) {
        const u32x4* s = (const u32x4*)(asrc + kn);
        ((u32x4*)(Als[cur ^ 1] + aoff))[0] = s[0];
        ((u32x4*)(Als[cur ^ 1] + aoff))[1] = s[1];
      }
      const float* brow = bsrc0 + (size_t)kn * D_OUT;
      if (kn + BK < D_HID) __builtin_prefetch(brow + (size_t)BK * D_OUT, 0, 0);
#pragma unroll
      for (int i = 0; i < 8; ++i)
        *(unsigned int*)(Bls[cur ^ 1] + (nsub + i) * BKP + kp) =
            pack2bf(brow[i], brow[i + D_OUT]);
    }
#pragma unroll
    for (int m = 0; m < 2; ++m) {
      const v16bf af = load_a_frag(Als[cur], mi0 + m, lane);
#pragma unroll
      for (int n = 0; n < 4; ++n) {
        const v16bf bfr = load_b_frag(Bls[cur], ni0 + n, lane);
        acc[m][n] = __builtin_amdgcn_wmma_f32_16x16x32_bf16(
            false, af, false, bfr, (short)0, acc[m][n], false, false);
      }
    }
    cur ^= 1;
  }

  const float* bias = b2 + (size_t)e * D_OUT + n0;
  const int lcol  = lane & 15;
  const int rhalf = (lane >> 4) * 8;
  float bv[4];
#pragma unroll
  for (int n = 0; n < 4; ++n) bv[n] = bias[(ni0 + n) * 16 + lcol];
#pragma unroll
  for (int m = 0; m < 2; ++m) {
#pragma unroll
    for (int v = 0; v < 8; ++v) {
      const int rloc = (mi0 + m) * 16 + rhalf + v;
      if (rloc < rows) {
        const int token = perm[row0 + rloc];
        const float s = scal[token];
        float* orow = out + (size_t)token * D_OUT + n0;
#pragma unroll
        for (int n = 0; n < 4; ++n)
          orow[(ni0 + n) * 16 + lcol] = (acc[m][n][v] + bv[n]) * s;
      }
    }
  }
}

extern "C" void kernel_launch(void* const* d_in, const int* in_sizes, int n_in,
                              void* d_out, int out_size, void* d_ws, size_t ws_size,
                              hipStream_t stream) {
  (void)in_sizes; (void)n_in; (void)out_size; (void)ws_size;
  const float* x  = (const float*)d_in[0];
  const float* Wr = (const float*)d_in[1];
  const float* br = (const float*)d_in[2];
  const float* W1 = (const float*)d_in[3];
  const float* b1 = (const float*)d_in[4];
  const float* W2 = (const float*)d_in[5];
  const float* b2 = (const float*)d_in[6];
  float* out = (float*)d_out;

  char* ws = (char*)d_ws;
  const size_t H_BYTES = (size_t)N_TOKENS * D_HID * sizeof(unsigned short); // 32 MB
  unsigned short* h = (unsigned short*)ws;
  int*   topi = (int*)  (ws + H_BYTES);
  float* tops = (float*)(ws + H_BYTES + 1 * 16384);
  int*   perm = (int*)  (ws + H_BYTES + 2 * 16384);
  float* scal = (float*)(ws + H_BYTES + 3 * 16384);
  int*   te   = (int*)  (ws + H_BYTES + 4 * 16384);
  int*   tr   = te + 64;
  int*   tn   = tr + 64;

  moe_router<<<N_TOKENS / 8, 256, 0, stream>>>(x, Wr, br, topi, tops);
  moe_sort<<<1, 256, 0, stream>>>(topi, tops, perm, scal, te, tr, tn);
  moe_ffn1<<<dim3(MAX_TILES, D_HID / BN), 256, 0, stream>>>(x, W1, b1, perm, te, tr, tn, h);
  moe_ffn2<<<dim3(MAX_TILES, D_OUT / BN), 256, 0, stream>>>(h, W2, b2, perm, scal, te, tr, tn, out);
}
